// RGLRU_27401891349142
// MI455X (gfx1250) — matrix operand order repaired and checked by reading, not compiled, not run on hardware
//
#include <hip/hip_runtime.h>

// RG-LRU forward for MI455X (gfx1250, wave32, WMMA).
// B=4, T=4096, W=1024, H=8, BW=128.

#define B_  4
#define T_  4096
#define W_  1024
#define H_  8
#define BW_ 128
#define ROWS_    (B_ * T_)      // 16384 flattened (b,t) rows
#define ROWTILE_ 16             // rows per wave (WMMA M) == scan chunk length
#define ROWBLK_  128            // rows per workgroup (8 waves)
#define CHUNK_   16
#define NCHUNK_  (ROWS_ / CHUNK_)   // 1024 chunks total
#define CPB_     (T_ / CHUNK_)      // 256 chunks per batch

typedef __attribute__((ext_vector_type(16))) _Float16 v16h;
typedef __attribute__((ext_vector_type(8)))  float    v8f;
typedef __attribute__((ext_vector_type(4)))  float    v4f;

// ---------------------------------------------------------------------------
// Kernel 1: block-diagonal gates via V_WMMA_F32_16X16X32_F16, fused epilogue
// (sigmoid / exp / sqrt gating), plus per-16-row-chunk scan aggregates
// (A = prod a_eff, S = local recurrence end) computed in registers.
// ---------------------------------------------------------------------------
__global__ __launch_bounds__(256) void rglru_gates_wmma(
    const float* __restrict__ x,        // [B,T,W]
    const int*   __restrict__ segpos,   // [B,T]
    const float* __restrict__ a_param,  // [W]
    const float* __restrict__ w_in,     // [H,BW,BW]
    const float* __restrict__ b_in,     // [H,BW]
    const float* __restrict__ w_a,      // [H,BW,BW]
    const float* __restrict__ b_a,      // [H,BW]
    float* __restrict__ nx_out,         // [B,T,W] -> d_out
    float* __restrict__ a_out,          // [B,T,W] -> ws
    float* __restrict__ aggA,           // [NCHUNK, W] -> ws
    float* __restrict__ aggS)           // [NCHUNK, W] -> ws
{
    // LDS: transposed f16 copies of this head's weights.
    __shared__ __align__(64) _Float16 lds[2 * BW_ * BW_];
    _Float16* ldsWx = lds;               // w_in^T (f16)
    _Float16* ldsWa = lds + BW_ * BW_;   // w_a^T  (f16)

    const int h      = blockIdx.y;
    const int rowblk = blockIdx.x * ROWBLK_;
    const int tid    = threadIdx.x;

    const float* winH = w_in + h * BW_ * BW_;
    const float* waH  = w_a  + h * BW_ * BW_;
    for (int idx = tid; idx < BW_ * BW_; idx += 256) {
        const int i = idx >> 7;
        const int j = idx & (BW_ - 1);
        ldsWx[j * BW_ + i] = (_Float16)winH[idx];
        ldsWa[j * BW_ + i] = (_Float16)waH[idx];
    }
    __syncthreads();

    const int wave = tid >> 5;
    const int lane = tid & 31;
    const int m    = lane & 15;
    const int hi   = lane >> 4;
    const int r0   = rowblk + wave * ROWTILE_;
    const int ch0  = h * BW_;
    const int g    = r0 >> 4;   // global chunk index

    // ---- Full A tile (16 rows x 128 K) as f16 fragments (contiguous k = 16*hi + e per 32-k block).
    v16h afrag[4];
    const float* xrow = x + (size_t)(r0 + m) * W_ + ch0;
#pragma unroll
    for (int s = 0; s < 4; ++s) {
        const float* plo = xrow + 32 * s + 16 * hi;
        const float* phi = plo + 8;
        const v4f a0 = *(const v4f*)(plo);
        const v4f a1 = *(const v4f*)(plo + 4);
        const v4f a2 = *(const v4f*)(phi);
        const v4f a3 = *(const v4f*)(phi + 4);
        v16h f;
#pragma unroll
        for (int e = 0; e < 4; ++e) {
            f[e]      = (_Float16)a0[e];
            f[4 + e]  = (_Float16)a1[e];
            f[8 + e]  = (_Float16)a2[e];
            f[12 + e] = (_Float16)a3[e];
        }
        afrag[s] = f;
    }

    // Reset factors per output row (segment_pos == 0 -> a_eff = 0).
    const int* ps = segpos + r0 + 8 * hi;
    float segf[8];
#pragma unroll
    for (int v = 0; v < 8; ++v) segf[v] = (ps[v] != 0) ? 1.0f : 0.0f;

    for (int jt = 0; jt < 8; ++jt) {
        const int j = jt * 16 + m;
        v8f accx = {};
        v8f acca = {};
#pragma unroll
        for (int s = 0; s < 4; ++s) {
            const int koff = 32 * s + 16 * hi;   // contiguous k = 16*hi + e, same as A
            const v16h bx = *(const v16h*)&ldsWx[j * BW_ + koff];
            const v16h ba = *(const v16h*)&ldsWa[j * BW_ + koff];
            accx = __builtin_amdgcn_wmma_f32_16x16x32_f16(
                false, afrag[s], false, bx, (short)0, accx, false, false);
            acca = __builtin_amdgcn_wmma_f32_16x16x32_f16(
                false, afrag[s], false, ba, (short)0, acca, false, false);
        }

        // ---- Epilogue. C/D layout: row = v + 8*hi, col = m.
        const float bxs = b_in[h * BW_ + j];
        const float bas = b_a[h * BW_ + j];
        const float nsp = -8.0f * __logf(1.0f + __expf(a_param[ch0 + j]));

        const size_t eo = (size_t)(r0 + 8 * hi) * W_ + ch0 + j;
        const float* px = x      + eo;
        float*       po = nx_out + eo;
        float*       pw = a_out  + eo;

        float p = 1.0f, s = 0.0f;   // chunk-aggregate accumulators (8 rows)
#pragma unroll
        for (int v = 0; v < 8; ++v) {
            const float yx = accx[v] + bxs;
            const float ya = acca[v] + bas;
            const float gx = __builtin_amdgcn_rcpf(1.0f + __expf(-yx));
            const float ga = __builtin_amdgcn_rcpf(1.0f + __expf(-ya));
            const float la = nsp * ga;                 // log_a
            const float a  = __expf(la);
            const float mu = __builtin_amdgcn_sqrtf(fmaxf(1.0f - a * a, 0.0f));
            const float nx = px[v * W_] * gx * mu;
            const float ae = a * segf[v];
            po[v * W_] = nx;
            pw[v * W_] = ae;
            s = fmaf(ae, s, nx);                       // local scan (t ascending)
            p *= ae;
        }
        // Combine the two 8-row halves into a 16-row chunk aggregate.
        const float p2 = __shfl_xor(p, 16, 32);
        const float s2 = __shfl_xor(s, 16, 32);
        if (hi == 0) {
            // this lane: rows 0..7 (lo); partner: rows 8..15 (hi)
            const float A16 = p * p2;
            const float S16 = fmaf(p2, s, s2);
            const size_t go = (size_t)g * W_ + ch0 + j;
            aggA[go] = A16;
            aggS[go] = S16;
        }
    }
}

// ---------------------------------------------------------------------------
// Kernel 2: per-(batch,channel) Hillis-Steele scan over the 256 chunk
// aggregates; writes the carry-in h0 for every chunk. Grid (W, B) x 256.
// ---------------------------------------------------------------------------
__global__ __launch_bounds__(256) void rglru_chunk_scan(
    const float* __restrict__ aggA,
    const float* __restrict__ aggS,
    float* __restrict__ h0)             // [NCHUNK, W]
{
    __shared__ float sA[CPB_];
    __shared__ float sS[CPB_];
    const int c  = threadIdx.x;     // chunk-in-batch 0..255
    const int ch = blockIdx.x;      // channel
    const int b  = blockIdx.y;      // batch
    const int g  = b * CPB_ + c;

    float A = aggA[(size_t)g * W_ + ch];
    float S = aggS[(size_t)g * W_ + ch];
    sA[c] = A;
    sS[c] = S;
#pragma unroll
    for (int off = 1; off < CPB_; off <<= 1) {
        __syncthreads();
        float pa = 1.0f, psv = 0.0f;
        if (c >= off) { pa = sA[c - off]; psv = sS[c - off]; }
        __syncthreads();
        if (c >= off) {
            S = fmaf(A, psv, S);    // compose: left segment, then current
            A = A * pa;
            sA[c] = A;
            sS[c] = S;
        }
    }
    __syncthreads();
    const float carry = (c == 0) ? 0.0f : sS[c - 1];
    h0[(size_t)g * W_ + ch] = carry;
}

// ---------------------------------------------------------------------------
// Kernel 3: apply carries — rerun the 16-step local recurrence per chunk with
// its carry-in, in place on d_out. Fully parallel, coalesced. Grid (4, 1024).
// ---------------------------------------------------------------------------
__global__ __launch_bounds__(256) void rglru_apply(
    const float* __restrict__ a_buf,    // a_eff
    const float* __restrict__ h0,       // carry-in per chunk
    float* __restrict__ out)            // in: normed_x, out: h
{
    const int ch = blockIdx.x * 256 + threadIdx.x;
    const int g  = blockIdx.y;
    const size_t base = (size_t)(g * CHUNK_) * W_ + ch;

    float h = h0[(size_t)g * W_ + ch];
    const float* pa = a_buf + base;
    float*       po = out   + base;
#pragma unroll
    for (int k = 0; k < CHUNK_; ++k) {
        const float a  = pa[k * W_];
        const float nx = po[k * W_];
        h = fmaf(a, h, nx);
        po[k * W_] = h;
    }
}

extern "C" void kernel_launch(void* const* d_in, const int* in_sizes, int n_in,
                              void* d_out, int out_size, void* d_ws, size_t ws_size,
                              hipStream_t stream) {
    const float* x       = (const float*)d_in[0];
    const int*   segpos  = (const int*)d_in[1];
    const float* a_param = (const float*)d_in[2];
    const float* w_in    = (const float*)d_in[3];
    const float* b_in    = (const float*)d_in[4];
    const float* w_a     = (const float*)d_in[5];
    const float* b_a     = (const float*)d_in[6];
    float* out = (float*)d_out;

    // Workspace layout (floats): a_eff | aggA | aggS | h0
    float* a_ws = (float*)d_ws;                       // ROWS_*W_   (67 MB)
    float* aggA = a_ws + (size_t)ROWS_ * W_;          // NCHUNK_*W_ (4 MB)
    float* aggS = aggA + (size_t)NCHUNK_ * W_;        // NCHUNK_*W_ (4 MB)
    float* h0   = aggS + (size_t)NCHUNK_ * W_;        // NCHUNK_*W_ (4 MB)

    dim3 grid1(ROWS_ / ROWBLK_, H_);                  // 128 x 8
    rglru_gates_wmma<<<grid1, 256, 0, stream>>>(
        x, segpos, a_param, w_in, b_in, w_a, b_a, out, a_ws, aggA, aggS);

    dim3 grid2(W_, B_);                               // 1024 x 4
    rglru_chunk_scan<<<grid2, 256, 0, stream>>>(aggA, aggS, h0);

    dim3 grid3(W_ / 256, NCHUNK_);                    // 4 x 1024
    rglru_apply<<<grid3, 256, 0, stream>>>(a_ws, h0, out);
}
